// AffineCoupling_41077067218946
// MI455X (gfx1250) — matrix operand order repaired and checked by reading, not compiled, not run on hardware
//
#include <hip/hip_runtime.h>
#include <hip/hip_bf16.h>

#define BB       8
#define TT       4096
#define CH       256
#define CH2      512
#define NCTX     80
#define CTXP     96
#define KW       864
#define NLAYERS  8

typedef __attribute__((ext_vector_type(16))) _Float16 v16h;
typedef __attribute__((ext_vector_type(8)))  _Float16 v8h;
typedef __attribute__((ext_vector_type(8)))  float    v8f;

union H16 { v16h v; struct { v8h lo, hi; } s; };

__device__ __forceinline__ v8f wmma_f16(v16h a, v16h b, v8f c) {
  // v_wmma_f32_16x16x32_f16: D = A(16x32) * B(32x16) + C(16x16 f32)
  return __builtin_amdgcn_wmma_f32_16x16x32_f16(
      /*neg_a=*/false, a, /*neg_b=*/false, b,
      /*c_mod=*/(short)0, c, /*reuse_a=*/false, /*reuse_b=*/false);
}

// ---------------------------------------------------------------------------
// Gate kernel: a = [in_w taps | cond_w] @ [x(t-d);x(t);x(t+d);ctx(t)] + bias
// acts = tanh(a_lo) * sigmoid(a_hi), stored f16 channel-contiguous (B,T,256).
// Workgroup: 512 threads = 16 waves; tile M=512 (full), N=32 columns.
// Wave w owns rows {16w..16w+15} and {256+16w..+15} (matched gate pair).
// Interior blocks (no boundary clipping possible) take a branchless path.
// ---------------------------------------------------------------------------
__global__ __launch_bounds__(512) void gate_kernel(
    const _Float16* __restrict__ x16t,   // (B,T,256) f16
    const _Float16* __restrict__ ctx16t, // (B,T,96)  f16
    const _Float16* __restrict__ wpk,    // (512,864) f16, this layer
    const float*    __restrict__ gbias,  // (512) = in_b + cond_b slice
    _Float16*       __restrict__ acts,   // (B,T,256) f16
    int dil)
{
  const int b  = blockIdx.y;
  const int t0 = blockIdx.x << 5;
  const int wv = threadIdx.x >> 5;
  const int ln = threadIdx.x & 31;
  const int lr = ln & 15;
  const int kA = (ln >> 4) << 3;   // A/C lane-half offset (rows / K-interleave)
  const int kB = (ln >> 4) << 4;   // unused (B loads use kA, same K order as A)

  const int mlo = wv << 4;
  const int mhi = 256 + mlo;

  v8f clo0 = {}, clo1 = {}, chi0 = {}, chi1 = {};
  const v8h z8 = {};

  const _Float16* wlo = wpk + (size_t)(mlo + lr) * KW + kA;
  const _Float16* whi = wpk + (size_t)(mhi + lr) * KW + kA;

  const bool interior = (t0 >= dil) && (t0 + 31 + dil < TT);

  if (interior) {
    // ---- fast path: no predication, pure base+imm-offset b128 loads ----
    const _Float16* pb = x16t + (((size_t)b * TT + (t0 + lr - dil)) << 8) + kA;
    const size_t dstep = (size_t)dil << 8;
    for (int tap = 0; tap < 3; ++tap) {
#pragma unroll
      for (int cc = 0; cc < 8; ++cc) {
        const int kg = (tap << 8) + (cc << 5);
        H16 afl, afh, bf0, bf1;
        afl.s.lo = *(const v8h*)(wlo + kg);
        afl.s.hi = *(const v8h*)(wlo + kg + 16);
        afh.s.lo = *(const v8h*)(whi + kg);
        afh.s.hi = *(const v8h*)(whi + kg + 16);
        bf0.s.lo = *(const v8h*)(pb + (cc << 5));
        bf0.s.hi = *(const v8h*)(pb + (cc << 5) + 16);
        bf1.s.lo = *(const v8h*)(pb + (cc << 5) + (16 << 8));
        bf1.s.hi = *(const v8h*)(pb + (cc << 5) + (16 << 8) + 16);
        clo0 = wmma_f16(afl.v, bf0.v, clo0);
        clo1 = wmma_f16(afl.v, bf1.v, clo1);
        chi0 = wmma_f16(afh.v, bf0.v, chi0);
        chi1 = wmma_f16(afh.v, bf1.v, chi1);
      }
      pb += dstep;
    }
  } else {
    // ---- boundary path: per-lane zero clipping ----
    for (int kk = 0; kk < 24; ++kk) {
      const int kg = kk << 5;
      const int tap = kg >> 8;            // 0,1,2
      const int ch  = (kg & 255) + kA;
      const int ts  = t0 + lr + (tap - 1) * dil;

      H16 afl, afh, bf0, bf1;
      afl.s.lo = *(const v8h*)(wlo + kg);
      afl.s.hi = *(const v8h*)(wlo + kg + 16);
      afh.s.lo = *(const v8h*)(whi + kg);
      afh.s.hi = *(const v8h*)(whi + kg + 16);

      if (ts >= 0 && ts < TT) {
        const _Float16* p = x16t + (((size_t)b * TT + ts) << 8) + ch;
        bf0.s.lo = *(const v8h*)p; bf0.s.hi = *(const v8h*)(p + 16);
      } else { bf0.s.lo = z8; bf0.s.hi = z8; }
      const int ts1 = ts + 16;
      if (ts1 >= 0 && ts1 < TT) {
        const _Float16* p = x16t + (((size_t)b * TT + ts1) << 8) + ch;
        bf1.s.lo = *(const v8h*)p; bf1.s.hi = *(const v8h*)(p + 16);
      } else { bf1.s.lo = z8; bf1.s.hi = z8; }

      clo0 = wmma_f16(afl.v, bf0.v, clo0);
      clo1 = wmma_f16(afl.v, bf1.v, clo1);
      chi0 = wmma_f16(afh.v, bf0.v, chi0);
      chi1 = wmma_f16(afh.v, bf1.v, chi1);
    }
  }

  // ---- context channels (zero-padded to 96, never OOB) ----
  {
    const _Float16* pc = ctx16t + ((size_t)b * TT + t0 + lr) * CTXP + kA;
#pragma unroll
    for (int cc = 0; cc < 3; ++cc) {
      const int kg = 768 + (cc << 5);
      H16 afl, afh, bf0, bf1;
      afl.s.lo = *(const v8h*)(wlo + kg);
      afl.s.hi = *(const v8h*)(wlo + kg + 16);
      afh.s.lo = *(const v8h*)(whi + kg);
      afh.s.hi = *(const v8h*)(whi + kg + 16);
      bf0.s.lo = *(const v8h*)(pc + (cc << 5));
      bf0.s.hi = *(const v8h*)(pc + (cc << 5) + 16);
      bf1.s.lo = *(const v8h*)(pc + (cc << 5) + 16 * CTXP);
      bf1.s.hi = *(const v8h*)(pc + (cc << 5) + 16 * CTXP + 16);
      clo0 = wmma_f16(afl.v, bf0.v, clo0);
      clo1 = wmma_f16(afl.v, bf1.v, clo1);
      chi0 = wmma_f16(afh.v, bf0.v, chi0);
      chi1 = wmma_f16(afh.v, bf1.v, chi1);
    }
  }

  float blv[8], bhv[8];
  *(float4*)(blv)     = *(const float4*)(gbias + mlo + kA);
  *(float4*)(blv + 4) = *(const float4*)(gbias + mlo + kA + 4);
  *(float4*)(bhv)     = *(const float4*)(gbias + mhi + kA);
  *(float4*)(bhv + 4) = *(const float4*)(gbias + mhi + kA + 4);

#pragma unroll
  for (int n = 0; n < 2; ++n) {
    const int t = t0 + (n << 4) + lr;
    v8f cl = n ? clo1 : clo0;
    v8f ch = n ? chi1 : chi0;
    v8h o;
#pragma unroll
    for (int j = 0; j < 8; ++j) {
      float a = cl[j] + blv[j];
      float g = ch[j] + bhv[j];
      float s = 1.0f / (1.0f + __expf(-g));
      o[j] = (_Float16)(tanhf(a) * s);
    }
    *(v8h*)(acts + (((size_t)b * TT + t) << 8) + mlo + kA) = o;
  }
}

// ---------------------------------------------------------------------------
// Res/skip kernel: rs = rs_w(512x256) @ acts + rs_b.
// !last: x += rs[:256] (fp32 + f16 mirror), out += rs[256:].
//  last: out += rs[:256].
// ---------------------------------------------------------------------------
__global__ __launch_bounds__(512) void rs_kernel(
    const _Float16* __restrict__ acts,   // (B,T,256) f16
    const _Float16* __restrict__ rsw,    // (512,256) f16
    const float*    __restrict__ rsb,    // (512)
    float*          __restrict__ x32t,   // (B,T,256) f32
    _Float16*       __restrict__ x16t,   // (B,T,256) f16
    float*          __restrict__ o32t,   // (B,T,256) f32 skip accumulator
    int last)
{
  const int b  = blockIdx.y;
  const int t0 = blockIdx.x << 5;
  const int wv = threadIdx.x >> 5;
  const int ln = threadIdx.x & 31;
  const int lr = ln & 15;
  const int kA = (ln >> 4) << 3;
  const int kB = (ln >> 4) << 4;

  const int mlo = wv << 4;
  const int mhi = 256 + mlo;

  v8f clo0 = {}, clo1 = {}, chi0 = {}, chi1 = {};

  const _Float16* wlo = rsw + (size_t)(mlo + lr) * CH + kA;
  const _Float16* whi = rsw + (size_t)(mhi + lr) * CH + kA;
  const _Float16* pb  = acts + (((size_t)b * TT + t0 + lr) << 8) + kA;

#pragma unroll
  for (int kk = 0; kk < 8; ++kk) {
    const int kg = kk << 5;

    H16 afl, bf0, bf1;
    afl.s.lo = *(const v8h*)(wlo + kg);
    afl.s.hi = *(const v8h*)(wlo + kg + 16);
    bf0.s.lo = *(const v8h*)(pb + kg);
    bf0.s.hi = *(const v8h*)(pb + kg + 16);
    bf1.s.lo = *(const v8h*)(pb + kg + (16 << 8));
    bf1.s.hi = *(const v8h*)(pb + kg + (16 << 8) + 16);

    clo0 = wmma_f16(afl.v, bf0.v, clo0);
    clo1 = wmma_f16(afl.v, bf1.v, clo1);
    if (!last) {
      H16 afh;
      afh.s.lo = *(const v8h*)(whi + kg);
      afh.s.hi = *(const v8h*)(whi + kg + 16);
      chi0 = wmma_f16(afh.v, bf0.v, chi0);
      chi1 = wmma_f16(afh.v, bf1.v, chi1);
    }
  }

  float blv[8], bhv[8];
  *(float4*)(blv)     = *(const float4*)(rsb + mlo + kA);
  *(float4*)(blv + 4) = *(const float4*)(rsb + mlo + kA + 4);
  if (!last) {
    *(float4*)(bhv)     = *(const float4*)(rsb + mhi + kA);
    *(float4*)(bhv + 4) = *(const float4*)(rsb + mhi + kA + 4);
  }

#pragma unroll
  for (int n = 0; n < 2; ++n) {
    const int t = t0 + (n << 4) + lr;
    const size_t cidx = (((size_t)b * TT + t) << 8) + mlo + kA;
    v8f cl = n ? clo1 : clo0;
    v8f ch = n ? chi1 : chi0;
    if (!last) {
      float xv[8]; v8h xo;
      *(float4*)(xv)     = *(const float4*)(x32t + cidx);
      *(float4*)(xv + 4) = *(const float4*)(x32t + cidx + 4);
#pragma unroll
      for (int j = 0; j < 8; ++j) { xv[j] += cl[j] + blv[j]; xo[j] = (_Float16)xv[j]; }
      *(float4*)(x32t + cidx)     = *(float4*)(xv);
      *(float4*)(x32t + cidx + 4) = *(float4*)(xv + 4);
      *(v8h*)(x16t + cidx) = xo;

      float ov[8];
      *(float4*)(ov)     = *(const float4*)(o32t + cidx);
      *(float4*)(ov + 4) = *(const float4*)(o32t + cidx + 4);
#pragma unroll
      for (int j = 0; j < 8; ++j) ov[j] += ch[j] + bhv[j];
      *(float4*)(o32t + cidx)     = *(float4*)(ov);
      *(float4*)(o32t + cidx + 4) = *(float4*)(ov + 4);
    } else {
      float ov[8];
      *(float4*)(ov)     = *(const float4*)(o32t + cidx);
      *(float4*)(ov + 4) = *(const float4*)(o32t + cidx + 4);
#pragma unroll
      for (int j = 0; j < 8; ++j) ov[j] += cl[j] + blv[j];
      *(float4*)(o32t + cidx)     = *(float4*)(ov);
      *(float4*)(o32t + cidx + 4) = *(float4*)(ov + 4);
    }
  }
}

// ---------------------------------------------------------------------------
// Start conv (K=4 -> 256) + zero skip accumulator. One thread = 8 channels @ t.
// ---------------------------------------------------------------------------
__global__ __launch_bounds__(256) void start_kernel(
    const float* __restrict__ fc, const float* __restrict__ sw,
    const float* __restrict__ sb, float* __restrict__ x32t,
    _Float16* __restrict__ x16t, float* __restrict__ o32t)
{
  const int idx = blockIdx.x * blockDim.x + threadIdx.x; // B*T*32
  const int g  = idx & 31;
  const int bt = idx >> 5;
  const int t  = bt & (TT - 1);
  const int b  = bt >> 12;

  float f0[4];
#pragma unroll
  for (int c = 0; c < 4; ++c) f0[c] = fc[((size_t)b * 8 + c) * TT + t];

  const int m0 = g << 3;
  float xv[8]; v8h xo;
#pragma unroll
  for (int j = 0; j < 8; ++j) {
    const int m = m0 + j;
    float a = sb[m];
#pragma unroll
    for (int c = 0; c < 4; ++c) a += sw[m * 4 + c] * f0[c];
    xv[j] = a; xo[j] = (_Float16)a;
  }
  const size_t cidx = ((size_t)bt << 8) + m0;
  *(float4*)(x32t + cidx)     = *(float4*)(xv);
  *(float4*)(x32t + cidx + 4) = *(float4*)(xv + 4);
  *(v8h*)(x16t + cidx) = xo;
  float4 z = {0.f, 0.f, 0.f, 0.f};
  *(float4*)(o32t + cidx)     = z;
  *(float4*)(o32t + cidx + 4) = z;
}

// ---------------------------------------------------------------------------
// Weight / context repack kernels
// ---------------------------------------------------------------------------
__global__ void prep_wpack(const float* __restrict__ in_w,
                           const float* __restrict__ cond_w,
                           _Float16* __restrict__ wpk)
{
  const size_t idx = (size_t)blockIdx.x * blockDim.x + threadIdx.x;
  if (idx >= (size_t)NLAYERS * CH2 * KW) return;
  const int    k  = (int)(idx % KW);
  const size_t lm = idx / KW;            // l*512 + m
  float v;
  if (k < 768) {
    const int tap = k >> 8;
    const int c   = k & 255;
    v = in_w[(lm * CH + c) * 3 + tap];
  } else {
    const int c = k - 768;
    v = (c < NCTX) ? cond_w[lm * NCTX + c] : 0.0f;
  }
  wpk[idx] = (_Float16)v;
}

__global__ void prep_rsw(const float* __restrict__ rs_w, _Float16* __restrict__ rsw)
{
  const size_t idx = (size_t)blockIdx.x * blockDim.x + threadIdx.x;
  if (idx >= (size_t)NLAYERS * CH2 * CH) return;
  rsw[idx] = (_Float16)rs_w[idx];
}

__global__ void prep_gbias(const float* __restrict__ in_b,
                           const float* __restrict__ cond_b,
                           float* __restrict__ gb)
{
  const int idx = blockIdx.x * blockDim.x + threadIdx.x;
  if (idx >= NLAYERS * CH2) return;
  gb[idx] = in_b[idx] + cond_b[idx];
}

__global__ void prep_ctx(const float* __restrict__ ctx, _Float16* __restrict__ c16)
{
  const int idx = blockIdx.x * blockDim.x + threadIdx.x; // B*T
  const int t = idx & (TT - 1);
  const int b = idx >> 12;
  _Float16 buf[CTXP];
#pragma unroll
  for (int c = 0; c < NCTX; ++c) buf[c] = (_Float16)ctx[((size_t)b * NCTX + c) * TT + t];
#pragma unroll
  for (int c = NCTX; c < CTXP; ++c) buf[c] = (_Float16)0.0f;
  v8h* dst = (v8h*)(c16 + (size_t)idx * CTXP);
#pragma unroll
  for (int i = 0; i < CTXP / 8; ++i) dst[i] = *(const v8h*)(buf + 8 * i);
}

// ---------------------------------------------------------------------------
// End conv (256 -> 8) + affine coupling + output assembly
// ---------------------------------------------------------------------------
__global__ __launch_bounds__(256) void end_kernel(
    const float* __restrict__ skip, const float* __restrict__ ew,
    const float* __restrict__ eb, const float* __restrict__ fc,
    float* __restrict__ out)
{
  const int idx = blockIdx.x * blockDim.x + threadIdx.x; // B*T
  const int t = idx & (TT - 1);
  const int b = idx >> 12;

  float acc[8];
#pragma unroll
  for (int o = 0; o < 8; ++o) acc[o] = eb[o];

  const float* sk = skip + ((size_t)idx << 8);
#pragma unroll 4
  for (int c = 0; c < CH; ++c) {
    const float s = sk[c];
#pragma unroll
    for (int o = 0; o < 8; ++o) acc[o] += ew[o * CH + c] * s;
  }

#pragma unroll
  for (int c = 0; c < 4; ++c) {
    const float f0v = fc[((size_t)b * 8 + c) * TT + t];
    out[((size_t)b * 8 + c) * TT + t] = f0v;                        // f0 passthrough
    const float f1v = fc[((size_t)b * 8 + 4 + c) * TT + t];
    const float ls  = acc[4 + c];
    out[((size_t)b * 8 + 4 + c) * TT + t] = __expf(ls) * f1v + acc[c];
    out[(size_t)BB * 8 * TT + ((size_t)b * 4 + c) * TT + t] = ls;   // log_s
  }
}

// ---------------------------------------------------------------------------
extern "C" void kernel_launch(void* const* d_in, const int* in_sizes, int n_in,
                              void* d_out, int out_size, void* d_ws, size_t ws_size,
                              hipStream_t stream)
{
  (void)in_sizes; (void)n_in; (void)out_size; (void)ws_size;

  const float* forecast = (const float*)d_in[0];
  const float* context  = (const float*)d_in[1];
  const float* start_w  = (const float*)d_in[2];
  const float* start_b  = (const float*)d_in[3];
  const float* cond_w   = (const float*)d_in[4];
  const float* cond_b   = (const float*)d_in[5];
  const float* in_w     = (const float*)d_in[6];
  const float* in_b     = (const float*)d_in[7];
  const float* rs_w     = (const float*)d_in[8];
  const float* rs_b     = (const float*)d_in[9];
  const float* end_w    = (const float*)d_in[10];
  const float* end_b    = (const float*)d_in[11];

  char* ws = (char*)d_ws;
  float*    x32t = (float*)ws;     ws += (size_t)BB * TT * CH * 4;
  float*    o32t = (float*)ws;     ws += (size_t)BB * TT * CH * 4;
  _Float16* x16t = (_Float16*)ws;  ws += (size_t)BB * TT * CH * 2;
  _Float16* a16t = (_Float16*)ws;  ws += (size_t)BB * TT * CH * 2;
  _Float16* c16t = (_Float16*)ws;  ws += (size_t)BB * TT * CTXP * 2;
  _Float16* wpk  = (_Float16*)ws;  ws += (size_t)NLAYERS * CH2 * KW * 2;
  _Float16* rsw  = (_Float16*)ws;  ws += (size_t)NLAYERS * CH2 * CH * 2;
  float*    gb   = (float*)ws;     ws += (size_t)NLAYERS * CH2 * 4;

  {
    const size_t n = (size_t)NLAYERS * CH2 * KW;
    prep_wpack<<<(unsigned)((n + 255) / 256), 256, 0, stream>>>(in_w, cond_w, wpk);
  }
  {
    const size_t n = (size_t)NLAYERS * CH2 * CH;
    prep_rsw<<<(unsigned)((n + 255) / 256), 256, 0, stream>>>(rs_w, rsw);
  }
  prep_gbias<<<(NLAYERS * CH2 + 255) / 256, 256, 0, stream>>>(in_b, cond_b, gb);
  prep_ctx<<<(BB * TT) / 128, 128, 0, stream>>>(context, c16t);
  start_kernel<<<(BB * TT * 32) / 256, 256, 0, stream>>>(forecast, start_w, start_b,
                                                         x32t, x16t, o32t);

  static const int DIL[NLAYERS] = {1, 2, 4, 8, 16, 32, 64, 128};
  dim3 gg(TT / 32, BB);
  for (int l = 0; l < NLAYERS; ++l) {
    gate_kernel<<<gg, 512, 0, stream>>>(x16t, c16t,
                                        wpk + (size_t)l * CH2 * KW,
                                        gb + l * CH2, a16t, DIL[l]);
    rs_kernel<<<gg, 512, 0, stream>>>(a16t,
                                      rsw + (size_t)l * CH2 * CH,
                                      rs_b + l * CH2,
                                      x32t, x16t, o32t,
                                      (l == NLAYERS - 1) ? 1 : 0);
  }

  end_kernel<<<(BB * TT) / 256, 256, 0, stream>>>(o32t, end_w, end_b, forecast,
                                                  (float*)d_out);
}
